// Seq2SeqLSTM_60696477827653
// MI455X (gfx1250) — matrix operand order repaired and checked by reading, not compiled, not run on hardware
//
#include <hip/hip_runtime.h>

typedef __bf16 bf16;
typedef bf16 v8bf  __attribute__((ext_vector_type(8)));
typedef bf16 v16bf __attribute__((ext_vector_type(16)));
typedef float v8f  __attribute__((ext_vector_type(8)));

#define B_  4096
#define T_  512
#define F_  16
#define H_  128
#define P_  96
#define NQ_ 3
#define G_  512   // 4*H

// padded LDS row strides (avoid 64-bank conflicts; keep 16B alignment)
#define W1W    136   // sW1 row stride (128 data + 8 pad) -> 272B = 17*16
#define ENC_AW 280   // encoder sA row stride (272 data + 8 pad) -> 560B = 35*16
#define DEC_AW 392   // decoder sA row stride (384 data + 8 pad) -> 784B = 49*16

// ---- WMMA wrapper: D = A(16xK=32 bf16) * B(32x16 bf16) + C(f32) ----
__device__ __forceinline__ v8f wmma_bf16(v16bf a, v16bf b, v8f c) {
  return __builtin_amdgcn_wmma_f32_16x16x32_bf16(false, a, false, b, (short)0, c, false, false);
}

// A-matrix fragment (16x32 bf16). rowp = &row[0] for row m=lane&15,
// kb = base + 32*chunk + (lane>=16 ? 16 : 0). e0..15 = K[kb..kb+15].
__device__ __forceinline__ v16bf loadA(const bf16* rowp, int kb) {
  v8bf lo = *(const v8bf*)(rowp + kb);
  v8bf hi = *(const v8bf*)(rowp + kb + 8);
  return __builtin_shufflevector(lo, hi, 0,1,2,3,4,5,6,7,8,9,10,11,12,13,14,15);
}

// B-matrix fragment (32x16 bf16), B[k][n] = W[n][k]. rowp = &W[n][0] for n=lane&15,
// kb = 32*chunk + (lane>=16 ? 16 : 0). 16 contiguous K values.
__device__ __forceinline__ v16bf loadB(const bf16* rowp, int kb) {
  v8bf lo = *(const v8bf*)(rowp + kb);
  v8bf hi = *(const v8bf*)(rowp + kb + 8);
  return __builtin_shufflevector(lo, hi, 0,1,2,3,4,5,6,7,8,9,10,11,12,13,14,15);
}

// branch-free activations (v_exp_f32 / v_rcp_f32, no exec-divergent slow paths)
__device__ __forceinline__ float sigf(float x) {
  return __fdividef(1.0f, 1.0f + __expf(-x));
}
__device__ __forceinline__ float tanh_fast(float x) {
  float t = __expf(-2.0f * fabsf(x));
  float r = __fdividef(1.0f - t, 1.0f + t);
  return copysignf(r, x);
}

// =====================================================================
// Encoder: 2-layer LSTM over T=512 steps. 16 batch rows per workgroup.
// Double-buffered activations -> only 2 barriers per timestep.
// Layer-0 weights [Wih0|Whh0|0] (K=160) in VGPRs (B-fragments, wave-private).
// Layer-1 weights: Wih1 (K 0..127) in LDS, Whh1 (K 128..255) in VGPRs.
// =====================================================================
__global__ __launch_bounds__(256, 2) void enc_kernel(
    const float* __restrict__ x,
    const float* __restrict__ Wih0, const float* __restrict__ Whh0,
    const float* __restrict__ bih0, const float* __restrict__ bhh0,
    const float* __restrict__ Wih1, const float* __restrict__ Whh1,
    const float* __restrict__ bih1, const float* __restrict__ bhh1,
    float* __restrict__ c0_out, float* __restrict__ c1_out,
    bf16* __restrict__ h0_out, bf16* __restrict__ h1_out)
{
  __shared__ bf16 sW1[G_ * W1W];         // Wih1, row-major [n][k], padded stride
  __shared__ bf16 sAb[2][16][ENC_AW];    // ping-pong: [x:0..15 | h0:16..143 | h1:144..271 | pad]

  const int tid  = threadIdx.x;
  const int lane = tid & 31;
  const int w    = tid >> 5;        // wave 0..7
  const int hl   = (lane >> 4) & 1; // lane half
  const int ln   = lane & 15;
  const int b0   = blockIdx.x * 16;

  // stage Wih1 -> LDS (bf16), padded rows
  for (int idx = tid; idx < G_ * H_; idx += 256)
    sW1[(idx >> 7) * W1W + (idx & 127)] = (bf16)Wih1[idx];
  // zero both activation buffers (initial h state = 0, pads finite)
  for (int idx = tid; idx < 2 * 16 * ENC_AW; idx += 256) (&sAb[0][0][0])[idx] = (bf16)0.0f;

  // ---- register-resident weights (B-fragment layout, wave-private N-columns) ----
  v16bf wL0[4][5];   // combined [Wih0 | Whh0 | 0], K = 160
  v16bf wL1[4][4];   // Whh1 (chunks 4..7 of combined layer-1 K=256)
  float bias0[4], bias1[4];
  #pragma unroll
  for (int i = 0; i < 4; i++) {
    const int n = (w + 8 * i) * 16 + ln;
    bias0[i] = bih0[n] + bhh0[n];
    bias1[i] = bih1[n] + bhh1[n];
    const float* wih = Wih0 + (size_t)n * F_;
    const float* whh = Whh0 + (size_t)n * H_;
    #pragma unroll
    for (int c = 0; c < 5; c++) {
      v16bf v;
      #pragma unroll
      for (int e = 0; e < 16; e++) {
        int k = 32 * c + hl * 16 + e;
        float val = (k < 16) ? wih[k] : ((k < 144) ? whh[k - 16] : 0.0f);
        v[e] = (bf16)val;
      }
      wL0[i][c] = v;
    }
    const float* w1 = Whh1 + (size_t)n * H_;
    #pragma unroll
    for (int c = 0; c < 4; c++) {
      v16bf v;
      #pragma unroll
      for (int e = 0; e < 16; e++) v[e] = (bf16)w1[32 * c + hl * 16 + e];
      wL1[i][c] = v;
    }
  }

  v8f c0t = {}, c1t = {};
  __syncthreads();

  const int ka = hl * 16;                      // A half-lane sub-offset
  const int r_st = tid >> 4, f_st = tid & 15;  // x staging coords
  const size_t xbase = ((size_t)(b0 + r_st)) * T_ * F_ + f_st;

  // ping-pong pointers: step t reads rd (x(t), h0(t-1), h1(t-1)), writes wr (h0(t), h1(t))
  bf16 (*rd)[ENC_AW] = sAb[1];
  bf16 (*wr)[ENC_AW] = sAb[0];

  float xr = x[xbase];   // x(0), staged at top of first iteration

  #pragma clang loop unroll(disable)
  for (int t = 0; t < T_; t++) {
    // stage x_t (preloaded last iteration); prefetch two ahead into WGP caches
    rd[r_st][f_st] = (bf16)xr;
    __builtin_prefetch(&x[xbase + (size_t)(t + 2) * F_], 0, 3);
    __syncthreads();                         // B1: x(t) + h1(t-1) visible

    const bf16* arow_rd = rd[ln];
    const bf16* arow_wr = wr[ln];

    // ---------- layer 0 : A = rd [x | h0 | zpad], K = 160 ----------
    v8f a0, a1, a2, a3;
    #pragma unroll
    for (int e = 0; e < 8; e++) { a0[e] = bias0[0]; a1[e] = bias0[1]; a2[e] = bias0[2]; a3[e] = bias0[3]; }
    #pragma unroll
    for (int c = 0; c < 5; c++) {
      v16bf a = loadA(arow_rd, 32 * c + ka);
      a0 = wmma_bf16(a, wL0[0][c], a0);
      a1 = wmma_bf16(a, wL0[1][c], a1);
      a2 = wmma_bf16(a, wL0[2][c], a2);
      a3 = wmma_bf16(a, wL0[3][c], a3);
    }
    v8f h0n;
    #pragma unroll
    for (int e = 0; e < 8; e++) {
      float cn = sigf(a1[e]) * c0t[e] + sigf(a0[e]) * tanh_fast(a2[e]);
      c0t[e] = cn;
      h0n[e] = sigf(a3[e]) * tanh_fast(cn);
    }
    // write h0(t) into the OTHER buffer (no readers -> no WAR barrier needed)
    {
      const int col = 16 + 16 * w + ln;
      #pragma unroll
      for (int v = 0; v < 8; v++) wr[v + 8 * hl][col] = (bf16)h0n[v];
    }
    __syncthreads();                         // B2: h0(t) visible

    // preload x(t+1); wait lands next iteration, hidden behind layer 1
    xr = x[xbase + (size_t)(t + 1 < T_ ? t + 1 : t) * F_];

    // ---------- layer 1 : K = 256 = [h0(t) from wr | h1(t-1) from rd] ----------
    #pragma unroll
    for (int e = 0; e < 8; e++) { a0[e] = bias1[0]; a1[e] = bias1[1]; a2[e] = bias1[2]; a3[e] = bias1[3]; }
    #pragma unroll
    for (int c = 0; c < 4; c++) {   // h0 chunk: A from wr, B from LDS
      v16bf a = loadA(arow_wr, 16 + 32 * c + ka);
      const int kb = 32 * c + hl * 16;
      a0 = wmma_bf16(a, loadB(&sW1[(size_t)((w +  0) * 16 + ln) * W1W], kb), a0);
      a1 = wmma_bf16(a, loadB(&sW1[(size_t)((w +  8) * 16 + ln) * W1W], kb), a1);
      a2 = wmma_bf16(a, loadB(&sW1[(size_t)((w + 16) * 16 + ln) * W1W], kb), a2);
      a3 = wmma_bf16(a, loadB(&sW1[(size_t)((w + 24) * 16 + ln) * W1W], kb), a3);
    }
    #pragma unroll
    for (int c = 0; c < 4; c++) {   // h1 chunk: A from rd, B from registers
      v16bf a = loadA(arow_rd, 144 + 32 * c + ka);
      a0 = wmma_bf16(a, wL1[0][c], a0);
      a1 = wmma_bf16(a, wL1[1][c], a1);
      a2 = wmma_bf16(a, wL1[2][c], a2);
      a3 = wmma_bf16(a, wL1[3][c], a3);
    }
    v8f h1n;
    #pragma unroll
    for (int e = 0; e < 8; e++) {
      float cn = sigf(a1[e]) * c1t[e] + sigf(a0[e]) * tanh_fast(a2[e]);
      c1t[e] = cn;
      h1n[e] = sigf(a3[e]) * tanh_fast(cn);
    }
    {
      const int col = 144 + 16 * w + ln;
      #pragma unroll
      for (int v = 0; v < 8; v++) wr[v + 8 * hl][col] = (bf16)h1n[v];
    }
    // swap buffers; next B1 publishes h1(t)
    bf16 (*tmp)[ENC_AW] = rd; rd = wr; wr = tmp;
  }
  __syncthreads();
  // after final swap, rd = buffer written at t = T-1

  // ---- write final state to workspace ----
  {
    const int col = 16 * w + ln;
    #pragma unroll
    for (int v = 0; v < 8; v++) {
      const int row = v + 8 * hl;
      c0_out[(size_t)(b0 + row) * H_ + col] = c0t[v];
      c1_out[(size_t)(b0 + row) * H_ + col] = c1t[v];
    }
  }
  for (int idx = tid; idx < 16 * H_; idx += 256) {
    const int r = idx >> 7, k = idx & 127;
    h0_out[(size_t)(b0 + r) * H_ + k] = rd[r][16 + k];
    h1_out[(size_t)(b0 + r) * H_ + k] = rd[r][144 + k];
  }
}

// =====================================================================
// Decoder: P=96 steps with median feedback.
// Layer-0 combined [Wih0|Whh0] (K=256) fully in VGPRs.
// Layer-1: Wih1 in LDS, Whh1 in VGPRs. Fused 16-thread output head.
// =====================================================================
__global__ __launch_bounds__(256, 2) void dec_kernel(
    const float* __restrict__ x,
    const float* __restrict__ Wih0, const float* __restrict__ Whh0,
    const float* __restrict__ bih0, const float* __restrict__ bhh0,
    const float* __restrict__ Wih1, const float* __restrict__ Whh1,
    const float* __restrict__ bih1, const float* __restrict__ bhh1,
    const float* __restrict__ inpW, const float* __restrict__ inpb,
    const float* __restrict__ outW, const float* __restrict__ outb,
    const float* __restrict__ c0_in, const float* __restrict__ c1_in,
    const bf16* __restrict__ h0_in, const bf16* __restrict__ h1_in,
    float* __restrict__ out)
{
  __shared__ bf16 sW1[G_ * W1W];      // dec_Wih1, padded stride
  __shared__ bf16 sA[16][DEC_AW];     // per-row: [d:0..127 | h0:128..255 | h1:256..383 | pad]
  __shared__ float sYp[16];
  __shared__ float sInpW[H_], sInpb[H_];
  __shared__ float sOutW[NQ_][H_];
  __shared__ float sOutb[NQ_];

  const int tid  = threadIdx.x;
  const int lane = tid & 31;
  const int w    = tid >> 5;
  const int hl   = (lane >> 4) & 1;
  const int ln   = lane & 15;
  const int b0   = blockIdx.x * 16;

  for (int idx = tid; idx < G_ * H_; idx += 256)
    sW1[(idx >> 7) * W1W + (idx & 127)] = (bf16)Wih1[idx];
  // load initial h0/h1 state; zero d region
  for (int idx = tid; idx < 16 * H_; idx += 256) {
    const int r = idx >> 7, k = idx & 127;
    sA[r][k]        = (bf16)0.0f;
    sA[r][128 + k]  = h0_in[(size_t)(b0 + r) * H_ + k];
    sA[r][256 + k]  = h1_in[(size_t)(b0 + r) * H_ + k];
  }
  if (tid < H_) { sInpW[tid] = inpW[tid]; sInpb[tid] = inpb[tid]; }
  if (tid < NQ_ * H_) sOutW[tid / H_][tid % H_] = outW[tid];
  if (tid < NQ_) sOutb[tid] = outb[tid];
  if (tid < 16) sYp[tid] = x[((size_t)(b0 + tid)) * T_ * F_ + (size_t)(T_ - 1) * F_]; // x[:, -1, 0]

  // register weights
  v16bf wL0[4][8];   // [Wih0 | Whh0], K = 256
  v16bf wL1[4][4];   // Whh1
  float bias0[4], bias1[4];
  #pragma unroll
  for (int i = 0; i < 4; i++) {
    const int n = (w + 8 * i) * 16 + ln;
    bias0[i] = bih0[n] + bhh0[n];
    bias1[i] = bih1[n] + bhh1[n];
    const float* wi = Wih0 + (size_t)n * H_;
    const float* wh = Whh0 + (size_t)n * H_;
    #pragma unroll
    for (int c = 0; c < 8; c++) {
      v16bf v;
      #pragma unroll
      for (int e = 0; e < 16; e++) {
        int k = 32 * c + hl * 16 + e;
        v[e] = (bf16)((k < 128) ? wi[k] : wh[k - 128]);
      }
      wL0[i][c] = v;
    }
    const float* w1 = Whh1 + (size_t)n * H_;
    #pragma unroll
    for (int c = 0; c < 4; c++) {
      v16bf v;
      #pragma unroll
      for (int e = 0; e < 16; e++) v[e] = (bf16)w1[32 * c + hl * 16 + e];
      wL1[i][c] = v;
    }
  }

  // load c state (same C-layout mapping the encoder used to store it)
  v8f c0t, c1t;
  {
    const int col = 16 * w + ln;
    #pragma unroll
    for (int v = 0; v < 8; v++) {
      const int row = v + 8 * hl;
      c0t[v] = c0_in[(size_t)(b0 + row) * H_ + col];
      c1t[v] = c1_in[(size_t)(b0 + row) * H_ + col];
    }
  }
  __syncthreads();

  const bf16* arow = sA[ln];
  const int ka = hl * 16;

  #pragma clang loop unroll(disable)
  for (int p = 0; p < P_; p++) {
    // d = yp * inp_W + inp_b   (16 rows x 128)
    {
      const int row = tid >> 4, hb = (tid & 15) * 8;
      const float yp = sYp[row];
      #pragma unroll
      for (int j = 0; j < 8; j++) sA[row][hb + j] = (bf16)(yp * sInpW[hb + j] + sInpb[hb + j]);
    }
    __syncthreads();

    // ---------- layer 0 : [d | h0], K = 256, all-register B ----------
    v8f a0, a1, a2, a3;
    #pragma unroll
    for (int e = 0; e < 8; e++) { a0[e] = bias0[0]; a1[e] = bias0[1]; a2[e] = bias0[2]; a3[e] = bias0[3]; }
    #pragma unroll
    for (int c = 0; c < 8; c++) {
      v16bf a = loadA(arow, 32 * c + ka);
      a0 = wmma_bf16(a, wL0[0][c], a0);
      a1 = wmma_bf16(a, wL0[1][c], a1);
      a2 = wmma_bf16(a, wL0[2][c], a2);
      a3 = wmma_bf16(a, wL0[3][c], a3);
    }
    v8f h0n;
    #pragma unroll
    for (int e = 0; e < 8; e++) {
      float cn = sigf(a1[e]) * c0t[e] + sigf(a0[e]) * tanh_fast(a2[e]);
      c0t[e] = cn;
      h0n[e] = sigf(a3[e]) * tanh_fast(cn);
    }
    __syncthreads();                 // readers of old h0 done
    {
      const int col = 128 + 16 * w + ln;
      #pragma unroll
      for (int v = 0; v < 8; v++) sA[v + 8 * hl][col] = (bf16)h0n[v];
    }
    __syncthreads();                 // new h0 visible

    // ---------- layer 1 : [h0 | h1], K = 256 ----------
    #pragma unroll
    for (int e = 0; e < 8; e++) { a0[e] = bias1[0]; a1[e] = bias1[1]; a2[e] = bias1[2]; a3[e] = bias1[3]; }
    #pragma unroll
    for (int c = 0; c < 4; c++) {
      v16bf a = loadA(arow, 128 + 32 * c + ka);
      const int kb = 32 * c + hl * 16;
      a0 = wmma_bf16(a, loadB(&sW1[(size_t)((w +  0) * 16 + ln) * W1W], kb), a0);
      a1 = wmma_bf16(a, loadB(&sW1[(size_t)((w +  8) * 16 + ln) * W1W], kb), a1);
      a2 = wmma_bf16(a, loadB(&sW1[(size_t)((w + 16) * 16 + ln) * W1W], kb), a2);
      a3 = wmma_bf16(a, loadB(&sW1[(size_t)((w + 24) * 16 + ln) * W1W], kb), a3);
    }
    #pragma unroll
    for (int c = 0; c < 4; c++) {
      v16bf a = loadA(arow, 128 + 32 * (c + 4) + ka);
      a0 = wmma_bf16(a, wL1[0][c], a0);
      a1 = wmma_bf16(a, wL1[1][c], a1);
      a2 = wmma_bf16(a, wL1[2][c], a2);
      a3 = wmma_bf16(a, wL1[3][c], a3);
    }
    v8f h1n;
    #pragma unroll
    for (int e = 0; e < 8; e++) {
      float cn = sigf(a1[e]) * c1t[e] + sigf(a0[e]) * tanh_fast(a2[e]);
      c1t[e] = cn;
      h1n[e] = sigf(a3[e]) * tanh_fast(cn);
    }
    __syncthreads();                 // readers of old h1 done
    {
      const int col = 256 + 16 * w + ln;
      #pragma unroll
      for (int v = 0; v < 8; v++) sA[v + 8 * hl][col] = (bf16)h1n[v];
    }
    __syncthreads();                 // new h1 visible

    // ---------- fused output head (16 threads): dots + sort + feedback ----------
    if (tid < 16) {
      float y0 = sOutb[0], y1 = sOutb[1], y2 = sOutb[2];
      #pragma unroll 16
      for (int k = 0; k < H_; k++) {
        const float hv = (float)sA[tid][256 + k];
        y0 += hv * sOutW[0][k];
        y1 += hv * sOutW[1][k];
        y2 += hv * sOutW[2][k];
      }
      const float lo  = fminf(y0, fminf(y1, y2));
      const float hi  = fmaxf(y0, fmaxf(y1, y2));
      const float mid = fmaxf(fminf(y0, y1), fminf(fmaxf(y0, y1), y2));
      const size_t o = ((size_t)(b0 + tid) * P_ + p) * NQ_;
      out[o] = lo; out[o + 1] = mid; out[o + 2] = hi;
      sYp[tid] = mid;
    }
    __syncthreads();                 // sYp visible for next step's d
  }
}

extern "C" void kernel_launch(void* const* d_in, const int* in_sizes, int n_in,
                              void* d_out, int out_size, void* d_ws, size_t ws_size,
                              hipStream_t stream) {
  const float* x        = (const float*)d_in[0];
  const float* eWih0    = (const float*)d_in[1];
  const float* eWhh0    = (const float*)d_in[2];
  const float* ebih0    = (const float*)d_in[3];
  const float* ebhh0    = (const float*)d_in[4];
  const float* eWih1    = (const float*)d_in[5];
  const float* eWhh1    = (const float*)d_in[6];
  const float* ebih1    = (const float*)d_in[7];
  const float* ebhh1    = (const float*)d_in[8];
  const float* dWih0    = (const float*)d_in[9];
  const float* dWhh0    = (const float*)d_in[10];
  const float* dbih0    = (const float*)d_in[11];
  const float* dbhh0    = (const float*)d_in[12];
  const float* dWih1    = (const float*)d_in[13];
  const float* dWhh1    = (const float*)d_in[14];
  const float* dbih1    = (const float*)d_in[15];
  const float* dbhh1    = (const float*)d_in[16];
  const float* inpW     = (const float*)d_in[17];
  const float* inpb     = (const float*)d_in[18];
  const float* outW     = (const float*)d_in[19];
  const float* outb     = (const float*)d_in[20];

  const size_t NH = (size_t)B_ * H_;
  float* wsf = (float*)d_ws;
  float* c0  = wsf;
  float* c1  = wsf + NH;
  bf16*  h0  = (bf16*)(wsf + 2 * NH);
  bf16*  h1  = h0 + NH;

  enc_kernel<<<B_ / 16, 256, 0, stream>>>(x, eWih0, eWhh0, ebih0, ebhh0,
                                          eWih1, eWhh1, ebih1, ebhh1,
                                          c0, c1, h0, h1);
  dec_kernel<<<B_ / 16, 256, 0, stream>>>(x, dWih0, dWhh0, dbih0, dbhh0,
                                          dWih1, dWhh1, dbih1, dbhh1,
                                          inpW, inpb, outW, outb,
                                          c0, c1, h0, h1, (float*)d_out);
}
